// Diffusion_79834852098619
// MI455X (gfx1250) — compile-verified, matrix operand order checked by reading, not run on hardware
//
#include <hip/hip_runtime.h>
#include <math.h>

typedef unsigned short u16;
typedef __attribute__((ext_vector_type(8)))  __bf16 v8bf;
typedef __attribute__((ext_vector_type(16))) __bf16 v16bf;
typedef __attribute__((ext_vector_type(8)))  float  v8f;

#define BM 128   // rows per workgroup tile
#define KC 64    // K chunk staged through LDS
#define PA 72    // LDS row pitch (elements, multiple of 8 -> 16B aligned rows)
#define HDIM 512

union FragU { v16bf v; v8bf h[2]; };

__device__ inline u16 f2bf(float f) {
  unsigned u = __float_as_uint(f);
  u += 0x7FFFu + ((u >> 16) & 1u);          // round-to-nearest-even
  return (u16)(u >> 16);
}

// mish(v) = v * tanh(softplus(v)); tanh(log(1+e^v)) = ((1+e^v)^2-1)/((1+e^v)^2+1)
__device__ inline float mishf(float v) {
  float p = __expf(v);
  float q = 1.f + p;
  float s = q * q;
  float t = (s - 1.f) / (s + 1.f);
  return (v > 15.f) ? v : v * t;
}

__device__ inline unsigned mix32(unsigned a) {
  a ^= a >> 16; a *= 0x7feb352dU;
  a ^= a >> 15; a *= 0x846ca68bU;
  a ^= a >> 16;
  return a;
}

// hash -> standard normal via Box-Muller (approximates jax.random.normal)
__device__ inline float gaussf(unsigned idx, unsigned salt) {
  unsigned base = idx * 2u + salt * 0x9E3779B9u;
  unsigned a = mix32(base + 1u);
  unsigned b = mix32(base + 0x632BE5ABu);
  float u1 = (float)(a >> 8) * (1.f / 16777216.f) + (0.5f / 16777216.f);
  float u2 = (float)(b >> 8) * (1.f / 16777216.f);
  float r = sqrtf(-2.f * logf(u1));
  return r * __cosf(6.2831853071795864f * u2);
}

// CDNA5 16-bit 16x32 fragment (A; B stored transposed so it's identical):
// lanes 0-15: K = k0+0..7 and k0+16..23 ; lanes 16-31: K = k0+8..15 and k0+24..31
__device__ inline v16bf load_frag(const u16* row, int k0, int lane) {
  FragU f;
  int c0 = k0 + ((lane >> 4) & 1) * 8;
  f.h[0] = *(const v8bf*)(row + c0);
  f.h[1] = *(const v8bf*)(row + c0 + 16);
  return f.v;
}

// Async 16B copy global -> LDS (ASYNCcnt-tracked). Low 32 bits of a flat
// pointer to __shared__ == LDS byte offset.
__device__ inline void async_copy16(void* lds_dst, const void* gsrc) {
  unsigned lo = (unsigned)(size_t)lds_dst;
  asm volatile("global_load_async_to_lds_b128 %0, %1, off"
               :: "v"(lo), "v"(gsrc) : "memory");
}

// dst[n][k] = bf16(src[k][n]) for k < Ksrc else 0  (weights -> [N][Kdst] bf16)
__global__ void convertT_kernel(const float* __restrict__ src, u16* __restrict__ dst,
                                int Ksrc, int Kdst, int N) {
  int idx = blockIdx.x * 256 + threadIdx.x;
  if (idx >= N * Kdst) return;
  int n = idx / Kdst, k = idx - n * Kdst;
  float v = (k < Ksrc) ? src[k * N + n] : 0.f;
  dst[idx] = f2bf(v);
}

__global__ void init_x_kernel(float* __restrict__ x, int n) {
  int i = blockIdx.x * 256 + threadIdx.x;
  if (i < n) x[i] = gaussf((unsigned)i, 0xD1FFu);
}

// assemble padded bf16 MLP input: cols 0-15 = x, 16-47 = temb(t), 48-175 = state, 176-191 = 0
__global__ void assemble_kernel(const float* __restrict__ x, const float* __restrict__ state,
                                u16* __restrict__ inbuf, float t, int total) {
  int idx = blockIdx.x * 256 + threadIdx.x;
  if (idx >= total) return;
  int row = idx / 192, col = idx - row * 192;
  float v;
  if (col < 16) {
    v = x[row * 16 + col];
  } else if (col < 48) {
    int k = (col - 16) & 15;
    float fr = __expf(-logf(10000.f) * (float)k / 15.f);
    float ang = t * fr;
    v = (col < 32) ? __sinf(ang) : __cosf(ang);
  } else if (col < 176) {
    v = state[row * 128 + (col - 48)];
  } else {
    v = 0.f;
  }
  inbuf[idx] = f2bf(v);
}

// ---- generic GEMM + mish: out[B][512] = mish(A[B][K] @ WT[512][K]^T + bias) ----
// Async-to-LDS double-buffered K chunks, bf16 WMMA 16x16x32.
__global__ __launch_bounds__(256)
void gemm_mish_kernel(const u16* __restrict__ A, const u16* __restrict__ WT,
                      const float* __restrict__ bias, u16* __restrict__ out, int K) {
  __shared__ __align__(16) u16 Ain[2][BM * PA];
  __shared__ __align__(16) u16 Bs[2][BM * PA];
  int tid = threadIdx.x, lane = tid & 31, wave = tid >> 5;
  int m0 = blockIdx.x * BM, n0 = blockIdx.y * BM;
  int nch = K / KC;

  v8f zero = {};
  v8f acc[8];
  for (int i = 0; i < 8; ++i) acc[i] = zero;

  // issue async copies for one K chunk into buffer `buf` (8 ops per thread)
  auto issue_tile = [&](int buf, int kc) {
#pragma unroll
    for (int q = 0; q < 4; ++q) {
      int cid = q * 256 + tid;
      int r = cid >> 3, c8 = (cid & 7) * 8;
      async_copy16(&Ain[buf][r * PA + c8], &A[(m0 + r) * K + kc + c8]);
      async_copy16(&Bs[buf][r * PA + c8], &WT[(n0 + r) * K + kc + c8]);
    }
  };

  issue_tile(0, 0);
  for (int c = 0; c < nch; ++c) {
    int buf = c & 1;
    if (c + 1 < nch) {
      issue_tile(buf ^ 1, (c + 1) * KC);
      asm volatile("s_wait_asynccnt 0x8" ::: "memory");  // chunk c landed; c+1 in flight
    } else {
      asm volatile("s_wait_asynccnt 0x0" ::: "memory");
    }
    __syncthreads();
    const u16* arow = &Ain[buf][(wave * 16 + (lane & 15)) * PA];
#pragma unroll
    for (int j = 0; j < KC; j += 32) {
      v16bf a = load_frag(arow, j, lane);
#pragma unroll
      for (int nt = 0; nt < 8; ++nt) {
        v16bf b = load_frag(&Bs[buf][(nt * 16 + (lane & 15)) * PA], j, lane);
        acc[nt] = __builtin_amdgcn_wmma_f32_16x16x32_bf16(false, a, false, b,
                                                          (short)0, acc[nt], false, false);
      }
    }
    __syncthreads();   // safe to overwrite this buffer next-next chunk
  }

  int rbase = wave * 16 + ((lane >> 4) << 3);
  int nlane = lane & 15;
#pragma unroll
  for (int nt = 0; nt < 8; ++nt) {
    int ncol = n0 + nt * 16 + nlane;
    float bv = bias[ncol];
#pragma unroll
    for (int r = 0; r < 8; ++r) {
      int row = m0 + rbase + r;
      out[row * HDIM + ncol] = f2bf(mishf(acc[nt][r] + bv));
    }
  }
}

// ---- layer 3 + diffusion update: eps = h2 @ w3 + b3 ; x' = c1*(srac*x - srm1*eps) + c2*x + sigma*z ----
__global__ __launch_bounds__(256)
void layer3_kernel(const u16* __restrict__ h2, const u16* __restrict__ w3T,
                   const float* __restrict__ b3, const float* __restrict__ xin,
                   float* __restrict__ xout,
                   float srac, float srm1, float c1, float c2, float sigma,
                   unsigned salt) {
  __shared__ __align__(16) u16 Ain[2][BM * PA];
  __shared__ __align__(16) u16 Ws[16 * 520];   // full w3T (16 x 512, padded pitch)
  int tid = threadIdx.x, lane = tid & 31, wave = tid >> 5;
  int m0 = blockIdx.x * BM;
  const int nch = HDIM / KC;

  // one-time async copy of w3T (16x512): 4 ops per thread
#pragma unroll
  for (int q = 0; q < 4; ++q) {
    int cid = q * 256 + tid;
    int r = cid >> 6, c8 = (cid & 63) * 8;
    async_copy16(&Ws[r * 520 + c8], &w3T[r * HDIM + c8]);
  }
  auto issue_a = [&](int buf, int kc) {
#pragma unroll
    for (int q = 0; q < 4; ++q) {
      int cid = q * 256 + tid;
      int r = cid >> 3, c8 = (cid & 7) * 8;
      async_copy16(&Ain[buf][r * PA + c8], &h2[(m0 + r) * HDIM + kc + c8]);
    }
  };

  v8f acc = {};
  issue_a(0, 0);
  for (int c = 0; c < nch; ++c) {
    int buf = c & 1;
    if (c + 1 < nch) {
      issue_a(buf ^ 1, (c + 1) * KC);
      asm volatile("s_wait_asynccnt 0x4" ::: "memory");  // Ws + chunk c landed
    } else {
      asm volatile("s_wait_asynccnt 0x0" ::: "memory");
    }
    __syncthreads();
    const u16* arow = &Ain[buf][(wave * 16 + (lane & 15)) * PA];
    const u16* brow = &Ws[(lane & 15) * 520];
#pragma unroll
    for (int j = 0; j < KC; j += 32) {
      v16bf a = load_frag(arow, j, lane);
      v16bf b = load_frag(brow, c * KC + j, lane);
      acc = __builtin_amdgcn_wmma_f32_16x16x32_bf16(false, a, false, b,
                                                    (short)0, acc, false, false);
    }
    __syncthreads();
  }

  int rbase = wave * 16 + ((lane >> 4) << 3);
  int nlane = lane & 15;
  float bv = b3[nlane];
#pragma unroll
  for (int r = 0; r < 8; ++r) {
    int row = m0 + rbase + r;
    float eps = acc[r] + bv;
    float xo = xin[row * 16 + nlane];
    float xrec = srac * xo - srm1 * eps;
    float mean = c1 * xrec + c2 * xo;
    float z = gaussf((unsigned)(row * 16 + nlane), salt);
    xout[row * 16 + nlane] = mean + sigma * z;
  }
}

extern "C" void kernel_launch(void* const* d_in, const int* in_sizes, int n_in,
                              void* d_out, int out_size, void* d_ws, size_t ws_size,
                              hipStream_t stream) {
  const float* state = (const float*)d_in[0];
  const float* w1 = (const float*)d_in[1];
  const float* b1 = (const float*)d_in[2];
  const float* w2 = (const float*)d_in[3];
  const float* b2 = (const float*)d_in[4];
  const float* w3 = (const float*)d_in[5];
  const float* b3 = (const float*)d_in[6];
  int B = in_sizes[0] / 128;

  char* p = (char*)d_ws;
  u16* w1T = (u16*)p; p += 512 * 192 * 2;
  u16* w2T = (u16*)p; p += 512 * 512 * 2;
  u16* w3T = (u16*)p; p += 16 * 512 * 2;
  u16* inb = (u16*)p; p += (size_t)B * 192 * 2;
  u16* h1  = (u16*)p; p += (size_t)B * HDIM * 2;
  u16* h2  = (u16*)p; p += (size_t)B * HDIM * 2;
  float* xb = (float*)p;

  convertT_kernel<<<(512 * 192 + 255) / 256, 256, 0, stream>>>(w1, w1T, 176, 192, 512);
  convertT_kernel<<<(512 * 512 + 255) / 256, 256, 0, stream>>>(w2, w2T, 512, 512, 512);
  convertT_kernel<<<(16 * 512 + 255) / 256, 256, 0, stream>>>(w3, w3T, 512, 512, 16);
  init_x_kernel<<<(B * 16 + 255) / 256, 256, 0, stream>>>(xb, B * 16);

  // diffusion schedule (host-side, deterministic; passed as kernel args)
  const int T = 10;
  double betas[10], alphas[10], acs[10], ac = 1.0;
  float srac[10], srm1[10], c1a[10], c2a[10], sg[10];
  for (int i = 0; i < T; ++i) {
    betas[i] = 1e-4 + (2e-2 - 1e-4) * (double)i / (double)(T - 1);
    alphas[i] = 1.0 - betas[i];
    ac *= alphas[i];
    acs[i] = ac;
  }
  for (int i = 0; i < T; ++i) {
    double a = acs[i];
    double ap = (i == 0) ? 1.0 : acs[i - 1];
    srac[i] = (float)sqrt(1.0 / a);
    srm1[i] = (float)sqrt(1.0 / a - 1.0);
    double pv = betas[i] * (1.0 - ap) / (1.0 - a);
    double lv = log(pv < 1e-20 ? 1e-20 : pv);
    sg[i] = (i > 0) ? (float)exp(0.5 * lv) : 0.f;
    c1a[i] = (float)(betas[i] * sqrt(ap) / (1.0 - a));
    c2a[i] = (float)((1.0 - ap) * sqrt(alphas[i]) / (1.0 - a));
  }

  dim3 gridG(B / BM, HDIM / BM);
  for (int i = T - 1; i >= 0; --i) {
    assemble_kernel<<<(B * 192 + 255) / 256, 256, 0, stream>>>(xb, state, inb, (float)i, B * 192);
    gemm_mish_kernel<<<gridG, 256, 0, stream>>>(inb, w1T, b1, h1, 192);
    gemm_mish_kernel<<<gridG, 256, 0, stream>>>(h1, w2T, b2, h2, 512);
    float* dst = (i == 0) ? (float*)d_out : xb;
    layer3_kernel<<<dim3(B / BM), 256, 0, stream>>>(h2, w3T, b3, xb, dst,
        srac[i], srm1[i], c1a[i], c2a[i], sg[i], (unsigned)i);
  }
}
